// APPNP_33861522161863
// MI455X (gfx1250) — matrix operand order repaired and checked by reading, not compiled, not run on hardware
//
#include <hip/hip_runtime.h>
#include <hip/hip_bf16.h>

typedef __attribute__((ext_vector_type(16))) __bf16    v16bf;
typedef __attribute__((ext_vector_type(8)))  float     v8f;
typedef __attribute__((ext_vector_type(8)))  unsigned  v8u;

#define NHID  256
#define NCLS  64
#define KIN   512

__device__ __forceinline__ unsigned short f2bf(float f) {
    __hip_bfloat16 t = __float2bfloat16(f);
    unsigned short r;
    __builtin_memcpy(&r, &t, sizeof(r));
    return r;
}
__device__ __forceinline__ unsigned packbf2(float a, float b) {
    __hip_bfloat162 t = __float22bfloat162_rn(make_float2(a, b));
    unsigned r;
    __builtin_memcpy(&r, &t, sizeof(r));
    return r;
}

// Fused MLP: h = relu(F@W1+b1)@W2+b2   (64 rows per block, 256 threads = 8 waves)
__global__ __launch_bounds__(256)
void appnp_mlp_kernel(const float* __restrict__ feat,
                      const float* __restrict__ W1, const float* __restrict__ b1,
                      const float* __restrict__ W2, const float* __restrict__ b2,
                      float* __restrict__ h, int N)
{
    __shared__ __align__(16) unsigned short As [64][32];    // F tile, row-major bf16
    __shared__ __align__(16) unsigned short BsT[256][32];   // W1 k-slice, transposed [n][k]
    __shared__ __align__(16) unsigned short Xs [64][256];   // relu(F@W1+b1), row-major bf16
    __shared__ __align__(16) unsigned short W2T[64][256];   // W2 transposed [n][k]
    __shared__ float b1s[256];
    __shared__ float b2s[64];

    const int tid  = threadIdx.x;
    const int wave = tid >> 5;
    const int lane = tid & 31;
    const int lm   = lane & 15;      // M (A) or N (B/C) index within a 16-tile
    const int hilo = lane >> 4;      // lane half
    const int row0 = blockIdx.x * 64;
    const bool full = (row0 + 64 <= N);

    // one-time staging: biases + W2 transposed (float4 global reads)
    b1s[tid] = b1[tid];
    if (tid < 64) b2s[tid] = b2[tid];
    #pragma unroll
    for (int i = 0; i < 16; ++i) {               // thread tid reads W2 row k=tid
        float4 v = *(const float4*)&W2[tid * 64 + i * 4];
        W2T[i * 4 + 0][tid] = f2bf(v.x);
        W2T[i * 4 + 1][tid] = f2bf(v.y);
        W2T[i * 4 + 2][tid] = f2bf(v.z);
        W2T[i * 4 + 3][tid] = f2bf(v.w);
    }

    // ---------------- GEMM1: 64x256 = F[64,512] @ W1[512,256], relu -> Xs ----
    const int rtile = wave & 3;                  // row tile (16 rows)
    const int nhalf = wave >> 2;                 // which 128-wide half of NHID
    v8f acc[8];
    #pragma unroll
    for (int j = 0; j < 8; ++j) acc[j] = (v8f)0.f;

    for (int kb = 0; kb < KIN / 32; ++kb) {
        __syncthreads();
        // stage A: 64x32 fp32 -> bf16, float4 loads, packed-dword LDS stores
        #pragma unroll
        for (int i = 0; i < 2; ++i) {
            int idx4 = i * 256 + tid;            // 512 float4s
            int r = idx4 >> 3, k4 = (idx4 & 7) * 4;
            float4 v = make_float4(0.f, 0.f, 0.f, 0.f);
            if (full || (row0 + r < N))
                v = *(const float4*)&feat[(size_t)(row0 + r) * KIN + kb * 32 + k4];
            *(uint2*)&As[r][k4] = make_uint2(packbf2(v.x, v.y), packbf2(v.z, v.w));
        }
        // stage B transposed: W1[kb*32+k][n] -> BsT[n][k], float4 global reads
        #pragma unroll
        for (int i = 0; i < 8; ++i) {
            int idx4 = i * 256 + tid;            // 2048 float4s
            int k = idx4 >> 6, n4 = (idx4 & 63) * 4;
            float4 v = *(const float4*)&W1[(size_t)(kb * 32 + k) * NHID + n4];
            BsT[n4 + 0][k] = f2bf(v.x);
            BsT[n4 + 1][k] = f2bf(v.y);
            BsT[n4 + 2][k] = f2bf(v.z);
            BsT[n4 + 3][k] = f2bf(v.w);
        }
        __syncthreads();

        // A fragment: 16x32 bf16, K packed 2-per-dword per ISA layout
        v8u au;
        #pragma unroll
        for (int v = 0; v < 8; ++v)
            au[v] = *(const unsigned*)&As[rtile * 16 + lm][(v < 4 ? 0 : 16) + hilo * 8 + (v & 3) * 2];
        v16bf af = __builtin_bit_cast(v16bf, au);

        #pragma unroll
        for (int j = 0; j < 8; ++j) {
            int n0 = nhalf * 128 + j * 16;
            v8u bu;
            #pragma unroll
            for (int v = 0; v < 8; ++v)
                bu[v] = *(const unsigned*)&BsT[n0 + lm][(v < 4 ? 0 : 16) + hilo * 8 + (v & 3) * 2];
            v16bf bf = __builtin_bit_cast(v16bf, bu);
            acc[j] = __builtin_amdgcn_wmma_f32_16x16x32_bf16(
                         false, af, false, bf, (short)0, acc[j], false, false);
        }
    }
    __syncthreads();
    // bias + relu -> Xs (bf16).  D layout: M = v + 8*hilo, N = lane%16
    #pragma unroll
    for (int j = 0; j < 8; ++j) {
        int n = nhalf * 128 + j * 16 + lm;
        #pragma unroll
        for (int v = 0; v < 8; ++v) {
            float x = acc[j][v] + b1s[n];
            Xs[rtile * 16 + v + 8 * hilo][n] = f2bf(x > 0.f ? x : 0.f);
        }
    }
    __syncthreads();

    // ---------------- GEMM2: 64x64 = Xs[64,256] @ W2[256,64] + b2 -> h -------
    #pragma unroll
    for (int t = 0; t < 2; ++t) {
        int tile = wave * 2 + t, rt = tile >> 2, ct = tile & 3;
        v8f c2 = (v8f)0.f;
        #pragma unroll
        for (int kb = 0; kb < NHID / 32; ++kb) {
            v8u au, bu;
            #pragma unroll
            for (int v = 0; v < 8; ++v) {
                au[v] = *(const unsigned*)&Xs[rt * 16 + lm][kb * 32 + (v < 4 ? 0 : 16) + hilo * 8 + (v & 3) * 2];
                bu[v] = *(const unsigned*)&W2T[ct * 16 + lm][kb * 32 + (v < 4 ? 0 : 16) + hilo * 8 + (v & 3) * 2];
            }
            c2 = __builtin_amdgcn_wmma_f32_16x16x32_bf16(
                     false, __builtin_bit_cast(v16bf, au), false,
                     __builtin_bit_cast(v16bf, bu), (short)0, c2, false, false);
        }
        int n = ct * 16 + lm;
        #pragma unroll
        for (int v = 0; v < 8; ++v) {
            int row = row0 + rt * 16 + v + 8 * hilo;
            if (full || row < N) h[(size_t)row * NCLS + n] = c2[v] + b2s[n];
        }
    }
}

// out = 0.1 * h   (float4 stream)
__global__ __launch_bounds__(256)
void appnp_init_kernel(const float* __restrict__ h, float* __restrict__ out, long long n4)
{
    long long i = (long long)blockIdx.x * blockDim.x + threadIdx.x;
    if (i < n4) {
        float4 v = ((const float4*)h)[i];
        v.x *= 0.1f; v.y *= 0.1f; v.z *= 0.1f; v.w *= 0.1f;
        ((float4*)out)[i] = v;
    }
}

// out[row] += 0.9*val*x[col] : 16 lanes per edge, float4 channels, native fp32 L2 atomics
__global__ __launch_bounds__(256)
void appnp_scatter_kernel(const float* __restrict__ vals,
                          const int* __restrict__ rows, const int* __restrict__ cols,
                          const float* __restrict__ x, float* __restrict__ out, int E)
{
    long long t = (long long)blockIdx.x * blockDim.x + threadIdx.x;
    int e = (int)(t >> 4);
    if (e >= E) return;
    __builtin_prefetch(&rows[e + 8192], 0, 0);   // stream edge lists ahead (global_prefetch_b8)
    int c = (int)(t & 15) * 4;
    int r = rows[e], cl = cols[e];
    float w = 0.9f * vals[e];
    float4 xv = *(const float4*)&x[(size_t)cl * NCLS + c];
    float* o = &out[(size_t)r * NCLS + c];
    __hip_atomic_fetch_add(o + 0, w * xv.x, __ATOMIC_RELAXED, __HIP_MEMORY_SCOPE_AGENT);
    __hip_atomic_fetch_add(o + 1, w * xv.y, __ATOMIC_RELAXED, __HIP_MEMORY_SCOPE_AGENT);
    __hip_atomic_fetch_add(o + 2, w * xv.z, __ATOMIC_RELAXED, __HIP_MEMORY_SCOPE_AGENT);
    __hip_atomic_fetch_add(o + 3, w * xv.w, __ATOMIC_RELAXED, __HIP_MEMORY_SCOPE_AGENT);
}

// row-wise log_softmax over 64 classes; one wave32 per row (2 values per lane)
__global__ __launch_bounds__(256)
void appnp_lsm_kernel(const float* __restrict__ x, float* __restrict__ out, int N)
{
    int wave = threadIdx.x >> 5, lane = threadIdx.x & 31;
    int row = blockIdx.x * 8 + wave;
    if (row >= N) return;
    const float* p = &x[(size_t)row * NCLS];
    float v0 = p[lane], v1 = p[lane + 32];
    float m = fmaxf(v0, v1);
    #pragma unroll
    for (int o = 16; o > 0; o >>= 1) m = fmaxf(m, __shfl_xor(m, o, 32));
    float s = __expf(v0 - m) + __expf(v1 - m);
    #pragma unroll
    for (int o = 16; o > 0; o >>= 1) s += __shfl_xor(s, o, 32);
    float ls = __logf(s);
    float* q = &out[(size_t)row * NCLS];
    q[lane]      = v0 - m - ls;
    q[lane + 32] = v1 - m - ls;
}

extern "C" void kernel_launch(void* const* d_in, const int* in_sizes, int n_in,
                              void* d_out, int out_size, void* d_ws, size_t ws_size,
                              hipStream_t stream)
{
    const float* feat = (const float*)d_in[0];
    const float* W1   = (const float*)d_in[1];
    const float* b1   = (const float*)d_in[2];
    const float* W2   = (const float*)d_in[3];
    const float* b2   = (const float*)d_in[4];
    const float* ev   = (const float*)d_in[5];
    const int*   er   = (const int*)d_in[6];
    const int*   ec   = (const int*)d_in[7];

    const int N = in_sizes[0] / KIN;
    const int E = in_sizes[5];

    float* h  = (float*)d_ws;                    // [N,64]
    float* xa = h  + (size_t)N * NCLS;           // ping
    float* xb = xa + (size_t)N * NCLS;           // pong

    appnp_mlp_kernel<<<(N + 63) / 64, 256, 0, stream>>>(feat, W1, b1, W2, b2, h, N);

    const long long n4 = (long long)N * NCLS / 4;
    const int initBlocks = (int)((n4 + 255) / 256);
    const long long scT  = (long long)E * 16;
    const int scBlocks   = (int)((scT + 255) / 256);

    const float* cur = h;
    float* nxt = xa;
    for (int k = 0; k < 10; ++k) {
        appnp_init_kernel<<<initBlocks, 256, 0, stream>>>(h, nxt, n4);
        appnp_scatter_kernel<<<scBlocks, 256, 0, stream>>>(ev, er, ec, cur, nxt, E);
        cur = nxt;
        nxt = (nxt == xa) ? xb : xa;
    }

    appnp_lsm_kernel<<<(N + 7) / 8, 256, 0, stream>>>(cur, (float*)d_out, N);
}
